// CuboidAttention_75960791597377
// MI455X (gfx1250) — 16-bit and 8-bit matrix operand order repaired and checked by reading, compiled, not run on hardware
//
#include <hip/hip_runtime.h>

#define DIM   256
#define NHEAD 8
#define DH    32
#define GTOK  8
#define SEQ   264
#define SEQP  272
#define MT_N  17
#define SCALEQ 0.17677669529663689f

typedef __attribute__((ext_vector_type(8)))  float  v8f;
typedef __attribute__((ext_vector_type(8)))  int    v8i;
typedef __attribute__((ext_vector_type(16))) __bf16 v16bf;

#define OFF_X    0
#define OFF_WT   69632
#define OFF_Q    94208
#define OFF_K    111616
#define OFF_V    129024
#define OFF_STG  147456
#define OFF_O    155648
#define SMEM_BYTES 294912
#define OFF_WB   0

__device__ __forceinline__ unsigned short f2bf(float f) {
  unsigned u = __builtin_bit_cast(unsigned, f);
  u += 0x7FFFu + ((u >> 16) & 1u);
  return (unsigned short)(u >> 16);
}

__device__ __forceinline__ unsigned f32_to_fp8_sw(float f) {
  unsigned u = __builtin_bit_cast(unsigned, f);
  unsigned sgn = (u >> 24) & 0x80u;
  unsigned ae = u & 0x7FFFFFFFu;
  if (ae >= 0x43E00000u) return sgn | 0x7Eu;
  if (ae <  0x3C800000u) return sgn;
  unsigned r = ae + 0x0007FFFFu + ((ae >> 20) & 1u);
  unsigned e = (r >> 23) - 120u;
  unsigned m = (r >> 20) & 7u;
  return sgn | (e << 3) | m;
}

__device__ __forceinline__ int pack4_fp8(float f0, float f1, float f2, float f3) {
#if __has_builtin(__builtin_amdgcn_cvt_pk_fp8_f32)
  int pk = __builtin_amdgcn_cvt_pk_fp8_f32(f0, f1, 0, false);
  pk = __builtin_amdgcn_cvt_pk_fp8_f32(f2, f3, pk, true);
  return pk;
#else
  return (int)(f32_to_fp8_sw(f0) | (f32_to_fp8_sw(f1) << 8) |
               (f32_to_fp8_sw(f2) << 16) | (f32_to_fp8_sw(f3) << 24));
#endif
}

__device__ __forceinline__ v8i load_a_fp8(const unsigned char* base, int stride,
                                          int row0, int k0, int lane) {
  const unsigned char* p = base + (row0 + (lane & 15)) * stride + k0 + ((lane >> 4) << 3);
  v8i a;
#pragma unroll
  for (int j = 0; j < 8; ++j)
    a[j] = *(const int*)(p + ((j & 1) << 2) + (((j >> 1) & 1) << 4) + (((j >> 2) & 1) << 5));
  return a;
}

__device__ __forceinline__ v8i load_b_fp8_nmajor(const unsigned char* base, int stride,
                                                 int n0, int k0, int lane) {
  const unsigned char* p = base + (n0 + (lane & 15)) * stride + k0 + ((lane >> 4) << 3);
  v8i b;
#pragma unroll
  for (int j = 0; j < 8; ++j)
    b[j] = *(const int*)(p + ((j & 1) << 2) + (((j >> 1) & 1) << 4) + (((j >> 2) & 1) << 5));
  return b;
}

__device__ __forceinline__ v8i load_a_bf16(const unsigned short* base, int strideE,
                                           int row0, int c0, int lane) {
  const unsigned short* p = base + (row0 + (lane & 15)) * strideE + c0 + ((lane >> 4) << 4);
  v8i a;
#pragma unroll
  for (int j = 0; j < 8; ++j)
    a[j] = *(const int*)(p + (j << 1));
  return a;
}

__device__ __forceinline__ v8i load_b_bf16_nmajor(const unsigned short* base, int strideE,
                                                  int n0, int c0, int lane) {
  const unsigned short* p = base + (n0 + (lane & 15)) * strideE + c0 + ((lane >> 4) << 4);
  v8i b;
#pragma unroll
  for (int j = 0; j < 8; ++j)
    b[j] = *(const int*)(p + (j << 1));
  return b;
}

__device__ __forceinline__ v8i load_b_bf16_kmajor(const unsigned short* base, int strideE,
                                                  int k0, int n0, int lane) {
  int n = n0 + (lane & 15);
  int kh = k0 + ((lane >> 4) << 4);
  v8i b;
#pragma unroll
  for (int j = 0; j < 8; ++j) {
    unsigned lo = base[(kh + 2 * j) * strideE + n];
    unsigned hi = base[(kh + 2 * j + 1) * strideE + n];
    b[j] = (int)(lo | (hi << 16));
  }
  return b;
}

__device__ __forceinline__ void store_c_bf16(unsigned short* base, int strideE,
                                             int row0, int c0, int lane, v8f c, float scl) {
  int col = c0 + (lane & 15);
  int rb = row0 + ((lane >> 4) << 3);
#pragma unroll
  for (int r = 0; r < 8; ++r)
    base[(rb + r) * strideE + col] = f2bf(c[r] * scl);
}

__device__ __forceinline__ v8f wmma_bf16(v8i a, v8i b, v8f c) {
  return __builtin_amdgcn_wmma_f32_16x16x32_bf16(
      false, __builtin_bit_cast(v16bf, a), false, __builtin_bit_cast(v16bf, b),
      (short)0, c, false, false);
}
__device__ __forceinline__ v8f wmma_fp8(v8i a, v8i b, v8f c) {
  return __builtin_amdgcn_wmma_f32_16x16x64_fp8_fp8(a, b, (short)0, c, false, false);
}

__global__ __launch_bounds__(256, 1)
void cuboid_attn(const float* __restrict__ x, const float* __restrict__ Wqkv,
                 const float* __restrict__ Wout, const float* __restrict__ bout,
                 const float* __restrict__ gtok, float* __restrict__ out) {
  extern __shared__ unsigned char smem[];
  unsigned char*  sX  = smem + OFF_X;
  unsigned char*  sWT = smem + OFF_WT;
  unsigned short* sQ  = (unsigned short*)(smem + OFF_Q);
  unsigned short* sK  = (unsigned short*)(smem + OFF_K);
  unsigned short* sV  = (unsigned short*)(smem + OFF_V);
  unsigned short* sSTG= (unsigned short*)(smem + OFF_STG);
  unsigned short* sO  = (unsigned short*)(smem + OFF_O);
  unsigned short* sWB = (unsigned short*)(smem + OFF_WB);

  const int tid = threadIdx.x, lane = tid & 31, wave = tid >> 5;
  const int bid = blockIdx.x;
  const int bb = bid >> 7;
  const int rr_ = bid & 127;
  const int tn = rr_ >> 6;
  const int hn = (rr_ >> 3) & 7;
  const int wn = rr_ & 7;
  const v8f Z = {0.f,0.f,0.f,0.f,0.f,0.f,0.f,0.f};

  for (int idx = tid; idx < SEQP * 64; idx += 256) {
    int row = idx >> 6;
    int c4 = (idx & 63) << 2;
    float f0 = 0.f, f1 = 0.f, f2 = 0.f, f3 = 0.f;
    if (row < GTOK) {
      const float4 v = *(const float4*)(gtok + row * DIM + c4);
      f0 = v.x; f1 = v.y; f2 = v.z; f3 = v.w;
    } else if (row < SEQ) {
      int i = row - GTOK;
      int ct = i >> 6, ch = (i >> 3) & 7, cw = i & 7;
      long off = ((((long)(bb * 8 + tn * 4 + ct) * 64 + (hn * 8 + ch)) * 64) +
                  (wn * 8 + cw)) * DIM + c4;
      const float4 v = *(const float4*)(x + off);
      f0 = v.x; f1 = v.y; f2 = v.z; f3 = v.w;
    }
    *(int*)(sX + row * DIM + c4) = pack4_fp8(f0, f1, f2, f3);
  }
  sV[272 * DH + tid] = 0;
  sV[272 * DH + 256 + tid] = 0;
  __syncthreads();

  for (int h = 0; h < NHEAD; ++h) {
    for (int e = tid; e < 96 * 64; e += 256) {
      int n = e >> 6;
      int k4 = (e & 63) << 2;
      int col = (n < 32) ? (h * DH + n)
              : (n < 64) ? (DIM + h * DH + (n - 32))
                         : (2 * DIM + h * DH + (n - 64));
      const float* wp = Wqkv + (long)k4 * 768 + col;
      *(int*)(sWT + n * DIM + k4) = pack4_fp8(wp[0], wp[768], wp[2 * 768], wp[3 * 768]);
    }
    __syncthreads();

    for (int mt = wave; mt < MT_N; mt += 8) {
      v8f acc[6] = {Z, Z, Z, Z, Z, Z};
#pragma unroll
      for (int kc = 0; kc < 4; ++kc) {
        v8i a = load_a_fp8(sX, DIM, mt * 16, kc * 64, lane);
#pragma unroll
        for (int nt = 0; nt < 6; ++nt) {
          v8i b = load_b_fp8_nmajor(sWT, DIM, nt * 16, kc * 64, lane);
          acc[nt] = wmma_fp8(a, b, acc[nt]);
        }
      }
      store_c_bf16(sQ, DH, mt * 16, 0,  lane, acc[0], SCALEQ);
      store_c_bf16(sQ, DH, mt * 16, 16, lane, acc[1], SCALEQ);
      store_c_bf16(sK, DH, mt * 16, 0,  lane, acc[2], 1.0f);
      store_c_bf16(sK, DH, mt * 16, 16, lane, acc[3], 1.0f);
      store_c_bf16(sV, DH, mt * 16, 0,  lane, acc[4], 1.0f);
      store_c_bf16(sV, DH, mt * 16, 16, lane, acc[5], 1.0f);
    }
    __syncthreads();

    unsigned short* stg = sSTG + wave * (16 * 32);
    for (int mt = wave; mt < MT_N; mt += 8) {
      v8i qa = load_a_bf16(sQ, DH, mt * 16, 0, lane);
      v8f sc[MT_N];
#pragma unroll
      for (int t = 0; t < MT_N; ++t) {
        v8i kb = load_b_bf16_nmajor(sK, DH, t * 16, 0, lane);
        sc[t] = wmma_bf16(qa, kb, Z);
      }
      if ((lane & 15) >= 8) {
#pragma unroll
        for (int r = 0; r < 8; ++r) sc[16][r] = -1e30f;
      }
      float inv[8];
#pragma unroll
      for (int r = 0; r < 8; ++r) {
        float m = -1e30f;
#pragma unroll
        for (int t = 0; t < MT_N; ++t) m = fmaxf(m, sc[t][r]);
#pragma unroll
        for (int sh = 1; sh < 16; sh <<= 1) m = fmaxf(m, __shfl_xor(m, sh, 16));
        float s = 0.f;
#pragma unroll
        for (int t = 0; t < MT_N; ++t) {
          float e = __expf(sc[t][r] - m);
          sc[t][r] = e;
          s += e;
        }
#pragma unroll
        for (int sh = 1; sh < 16; sh <<= 1) s += __shfl_xor(s, sh, 16);
        inv[r] = 1.0f / s;
      }
      v8f o0 = Z, o1 = Z;
#pragma unroll
      for (int kc = 0; kc < 9; ++kc) {
        asm volatile("" ::: "memory");
        __builtin_amdgcn_wave_barrier();
#pragma unroll
        for (int tt = 0; tt < 2; ++tt) {
          int tile = kc * 2 + tt;
          int col = (tt << 4) + (lane & 15);
          int rb = (lane >> 4) << 3;
#pragma unroll
          for (int r = 0; r < 8; ++r) {
            float pv = (tile < MT_N) ? sc[tile][r] * inv[r] : 0.f;
            stg[(rb + r) * 32 + col] = f2bf(pv);
          }
        }
        asm volatile("" ::: "memory");
        __builtin_amdgcn_wave_barrier();
        v8i pa  = load_a_bf16(stg, 32, 0, 0, lane);
        v8i vb0 = load_b_bf16_kmajor(sV, DH, kc * 32, 0,  lane);
        v8i vb1 = load_b_bf16_kmajor(sV, DH, kc * 32, 16, lane);
        o0 = wmma_bf16(pa, vb0, o0);
        o1 = wmma_bf16(pa, vb1, o1);
      }
      store_c_bf16(sO, DIM, mt * 16, h * DH,      lane, o0, 1.0f);
      store_c_bf16(sO, DIM, mt * 16, h * DH + 16, lane, o1, 1.0f);
    }
    __syncthreads();
  }

  for (int e = tid; e < DIM * DIM; e += 256) sWB[e] = f2bf(Wout[e]);
  __syncthreads();

  for (int mt = wave; mt < MT_N; mt += 8) {
    v8f acc[16];
#pragma unroll
    for (int nt = 0; nt < 16; ++nt) acc[nt] = Z;
#pragma unroll
    for (int kc = 0; kc < 8; ++kc) {
      v8i a = load_a_bf16(sO, DIM, mt * 16, kc * 32, lane);
#pragma unroll
      for (int nt = 0; nt < 16; ++nt) {
        v8i b = load_b_bf16_kmajor(sWB, DIM, kc * 32, nt * 16, lane);
        acc[nt] = wmma_bf16(a, b, acc[nt]);
      }
    }
    int colL = lane & 15;
    int rb = (lane >> 4) << 3;
#pragma unroll
    for (int nt = 0; nt < 16; ++nt) {
      int col = nt * 16 + colL;
      float bo = bout[col];
#pragma unroll
      for (int r = 0; r < 8; ++r) {
        int row = mt * 16 + rb + r;
        if (row >= GTOK && row < SEQ) {
          int i = row - GTOK;
          int ct = i >> 6, ch = (i >> 3) & 7, cw = i & 7;
          long off = ((((long)(bb * 8 + tn * 4 + ct) * 64 + (hn * 8 + ch)) * 64) +
                      (wn * 8 + cw)) * DIM + col;
          out[off] = acc[nt][r] + bo;
        }
      }
    }
  }
}

extern "C" void kernel_launch(void* const* d_in, const int* in_sizes, int n_in,
                              void* d_out, int out_size, void* d_ws, size_t ws_size,
                              hipStream_t stream) {
  (void)in_sizes; (void)n_in; (void)out_size; (void)d_ws; (void)ws_size;
  const float* x    = (const float*)d_in[0];
  const float* Wqkv = (const float*)d_in[1];
  const float* Wout = (const float*)d_in[2];
  const float* bout = (const float*)d_in[3];
  const float* gtok = (const float*)d_in[4];
  float* out = (float*)d_out;
  hipLaunchKernelGGL(cuboid_attn, dim3(256), dim3(256), SMEM_BYTES, stream,
                     x, Wqkv, Wout, bout, gtok, out);
}
